// CapsuleNetwork_64398739636940
// MI455X (gfx1250) — matrix operand order repaired and checked by reading, not compiled, not run on hardware
//
#include <hip/hip_runtime.h>
#include <hip/hip_bf16.h>
#include <math.h>

typedef __attribute__((ext_vector_type(16))) __bf16 v16bf;
typedef __attribute__((ext_vector_type(8)))  float  v8f;

#define B_  256
#define S_  64
#define E_  300
#define EP_ 320      // E padded to multiple of 32 (zero-filled)
#define H_  512
#define DA_ 350
#define DAP_ 384     // DA rows padded to multiple of 64 (garbage rows OK: not stored)
#define R_  8
#define C_  150
#define P_  16
#define CP_ (C_*P_)      // 2400
#define CPP_ 2432        // padded to multiple of 64

__device__ __forceinline__ unsigned short f2bf(float x) {
  union { float f; unsigned u; } t; t.f = x;
  unsigned r = t.u + 0x7FFFu + ((t.u >> 16) & 1u);
  return (unsigned short)(r >> 16);
}
__device__ __forceinline__ float bf2f(unsigned short h) {
  union { float f; unsigned u; } t; t.u = ((unsigned)h) << 16; return t.f;
}
__device__ __forceinline__ float sigf(float x) { return 1.0f / (1.0f + expf(-x)); }

// ---------------------------------------------------------------------------
// bf16 WMMA GEMM:  C[M,N] = A[M,K]*B[N,K]^T  (+ optional fused second segment
// A2[M,K2]*B2[N,K2]^T accumulated in registers).
// REQUIREMENTS (guaranteed by caller via padded scratch):
//   - K, K2 multiples of 32; M multiple of 64
//   - A has >= M rows; B has >= ceil64(N) rows; rows 16B-aligned
// Workgroup = 128 threads (4 waves), tile 64(M) x 64(N), K-step 32,
// software-pipelined: next K-tile global loads issued under current WMMAs.
// act: 0 = store f32, 1 = store bf16(tanh(x)).
// ---------------------------------------------------------------------------
__global__ __launch_bounds__(128) void k_gemm(
    const unsigned short* __restrict__ A,  long lda,  long aZ,
    const unsigned short* __restrict__ B,  long ldb,  long bZ,
    const unsigned short* __restrict__ A2, long lda2,
    const unsigned short* __restrict__ B2, long ldb2, int K2,
    void* __restrict__ C, long ldc, long cZ,
    int M, int N, int K, int act)
{
  __shared__ __align__(16) unsigned short sA[64][32];   // [m][k]
  __shared__ __align__(16) unsigned short sB[64][32];   // [n][k]

  const int z = blockIdx.z;
  A += (long)z * aZ;
  B += (long)z * bZ;
  float*          Cf = (float*)C          + (long)z * cZ;
  unsigned short* Cb = (unsigned short*)C + (long)z * cZ;

  const int tm = blockIdx.y * 64, tn = blockIdx.x * 64;
  const int wave = threadIdx.x >> 5, lane = threadIdx.x & 31;
  const int lm = lane & 15, lh = lane >> 4;

  v8f acc[4];
  for (int j = 0; j < 4; ++j)
    for (int v = 0; v < 8; ++v) acc[j][v] = 0.0f;

  // staging geometry: 64 rows; each thread covers one row-half (16 elems = 2x uint4)
  const int srow = threadIdx.x >> 1;          // 0..63
  const int sc8  = (threadIdx.x & 1) * 16;    // element offset 0 or 16

  const int nSeg = (A2 != nullptr) ? 2 : 1;
  for (int seg = 0; seg < nSeg; ++seg) {
    const unsigned short* aRow = (seg ? A2 : A) + (long)(tm + srow) * (seg ? lda2 : lda) + sc8;
    const unsigned short* bRow = (seg ? B2 : B) + (long)(tn + srow) * (seg ? ldb2 : ldb) + sc8;
    const int Ks = seg ? K2 : K;

    // preload first K-tile into registers
    uint4 ra0 = *(const uint4*)(aRow);
    uint4 ra1 = *(const uint4*)(aRow + 8);
    uint4 rb0 = *(const uint4*)(bRow);
    uint4 rb1 = *(const uint4*)(bRow + 8);

    for (int k0 = 0; k0 < Ks; k0 += 32) {
      // ---- commit staged registers to LDS ----
      *(uint4*)&sA[srow][sc8]     = ra0;
      *(uint4*)&sA[srow][sc8 + 8] = ra1;
      *(uint4*)&sB[srow][sc8]     = rb0;
      *(uint4*)&sB[srow][sc8 + 8] = rb1;
      __syncthreads();

      // ---- issue next tile's global loads (waits deferred past the WMMAs) ----
      if (k0 + 32 < Ks) {
        ra0 = *(const uint4*)(aRow + k0 + 32);
        ra1 = *(const uint4*)(aRow + k0 + 40);
        rb0 = *(const uint4*)(bRow + k0 + 32);
        rb1 = *(const uint4*)(bRow + k0 + 40);
        if (k0 + 64 < Ks) {   // speculative prefetch one more K-step out
          __builtin_prefetch(aRow + k0 + 64, 0, 3);
          __builtin_prefetch(bRow + k0 + 64, 0, 3);
        }
      }

      // ---- A fragment: lane 0-15 -> M=lane, K in {0..7}U{16..23};
      //                  lanes 16-31 -> K in {8..15}U{24..31}
      union { v16bf v; unsigned short u[16]; } af;
      int am = wave * 16 + lm;
#pragma unroll
      for (int i = 0; i < 8; ++i) {
        int kk = ((i < 4) ? (2 * i) : (16 + 2 * (i - 4))) + 8 * lh;
        af.u[2 * i]     = sA[am][kk];
        af.u[2 * i + 1] = sA[am][kk + 1];
      }
#pragma unroll
      for (int j = 0; j < 4; ++j) {
        // B fragment: lane 0-15 -> N=lane, K in {0..7}U{16..23}; lanes 16-31 -> K in {8..15}U{24..31}
        union { v16bf v; unsigned short u[16]; } bfrag;
        int bn = j * 16 + lm;
#pragma unroll
        for (int i = 0; i < 8; ++i) {
          int kk = ((i < 4) ? (2 * i) : (16 + 2 * (i - 4))) + 8 * lh;
          bfrag.u[2 * i]     = sB[bn][kk];
          bfrag.u[2 * i + 1] = sB[bn][kk + 1];
        }
        acc[j] = __builtin_amdgcn_wmma_f32_16x16x32_bf16(
            false, af.v, false, bfrag.v, (short)0, acc[j], false, false);
      }
      __syncthreads();
    }
  }

  // C/D layout: lane (0-15)->N, VGPR v -> M=v (+8 for lanes 16-31)
  for (int j = 0; j < 4; ++j)
    for (int v = 0; v < 8; ++v) {
      int gm = tm + wave * 16 + v + 8 * lh;
      int gn = tn + j * 16 + lm;
      if (gn < N) {
        if (act == 0) Cf[(long)gm * ldc + gn] = acc[j][v];
        else          Cb[(long)gm * ldc + gn] = f2bf(tanhf(acc[j][v]));
      }
    }
}

// ---------------------------------------------------------------------------
// dst[r][k] (rows x Kp, bf16) = src[r][k] (rows x K, f32); K..Kp-1 zero-filled
__global__ void k_cvt2d(const float* __restrict__ src, unsigned short* __restrict__ dst,
                        long rows, int K, int Kp) {
  long idx = (long)blockIdx.x * blockDim.x + threadIdx.x;
  if (idx >= rows * Kp) return;
  int k = (int)(idx % Kp);
  long r = idx / Kp;
  dst[idx] = (k < K) ? f2bf(src[r * K + k]) : (unsigned short)0;
}

// capsT[r][n][k] = bf16(capsW[r][k][n]); dst rows padded to CPP_ (pad rows unwritten)
__global__ void k_cvtT(const float* __restrict__ src, unsigned short* __restrict__ dst) {
  long idx = (long)blockIdx.x * blockDim.x + threadIdx.x;
  if (idx >= (long)R_ * CP_ * (2 * H_)) return;
  int k = (int)(idx % (2 * H_));
  long rn = idx / (2 * H_);
  int n = (int)(rn % CP_);
  int r = (int)(rn / CP_);
  dst[((long)r * CPP_ + n) * (2 * H_) + k] =
      f2bf(src[((long)r * (2 * H_) + k) * CP_ + n]);
}

// emb[s][b][e] = bf16(embedding[input_x[b][s]][e]); e in [E_,EP_) -> 0
__global__ void k_embed(const int* __restrict__ ix, const float* __restrict__ table,
                        unsigned short* __restrict__ embbf) {
  long idx = (long)blockIdx.x * blockDim.x + threadIdx.x;
  if (idx >= (long)S_ * B_ * EP_) return;
  int e = (int)(idx % EP_);
  long sb = idx / EP_;
  int bb = (int)(sb % B_);
  int ss = (int)(sb / B_);
  unsigned short v = 0;
  if (e < E_) {
    int tok = ix[bb * S_ + ss];
    v = f2bf(table[(long)tok * E_ + e]);
  }
  embbf[idx] = v;
}

// LSTM gate nonlinearity + state update; grid.z = direction
__global__ __launch_bounds__(256) void k_gates(
    const float* __restrict__ zbuf, float* __restrict__ cbuf,
    unsigned short* __restrict__ hbf, unsigned short* __restrict__ outbf,
    const float* __restrict__ bihf, const float* __restrict__ bhhf,
    const float* __restrict__ bihb, const float* __restrict__ bhhb, int t)
{
  int idx = blockIdx.x * blockDim.x + threadIdx.x;
  if (idx >= B_ * H_) return;
  int dir = blockIdx.z;
  int b = idx / H_, j = idx % H_;
  int s = dir ? (S_ - 1 - t) : t;
  const float* z  = zbuf + (long)dir * B_ * 4 * H_ + (long)b * 4 * H_;
  const float* bi = dir ? bihb : bihf;
  const float* bh = dir ? bhhb : bhhf;
  float zi = z[j]          + bi[j]          + bh[j];
  float zf = z[H_ + j]     + bi[H_ + j]     + bh[H_ + j];
  float zg = z[2 * H_ + j] + bi[2 * H_ + j] + bh[2 * H_ + j];
  float zo = z[3 * H_ + j] + bi[3 * H_ + j] + bh[3 * H_ + j];
  long ci = (long)dir * B_ * H_ + (long)b * H_ + j;
  float cc = cbuf[ci];
  cc = sigf(zf) * cc + sigf(zi) * tanhf(zg);
  float hh = sigf(zo) * tanhf(cc);
  cbuf[ci] = cc;
  hbf[ci] = f2bf(hh);
  outbf[((long)b * S_ + s) * (2 * H_) + dir * H_ + j] = f2bf(hh);
}

// att[b][r][s] = sum_a pre[b][s][a] * ws2[r][a]
__global__ void k_att(const unsigned short* __restrict__ prebf,
                      const float* __restrict__ ws2, float* __restrict__ att) {
  int idx = blockIdx.x * blockDim.x + threadIdx.x;
  if (idx >= B_ * R_ * S_) return;
  int s = idx % S_;
  int r = (idx / S_) % R_;
  int b = idx / (S_ * R_);
  const unsigned short* p = prebf + ((long)b * S_ + s) * DA_;
  const float* w = ws2 + r * DA_;
  float acc = 0.0f;
  for (int a = 0; a < DA_; ++a) acc += bf2f(p[a]) * w[a];
  att[((long)b * R_ + r) * S_ + s] = acc;
}

// softmax over S=64 per (b,r); block of 64 threads
__global__ __launch_bounds__(64) void k_softmax64(const float* __restrict__ att,
                                                  float* __restrict__ out) {
  __shared__ float red[64];
  long row = blockIdx.x;
  int t = threadIdx.x;
  float v = att[row * S_ + t];
  red[t] = v; __syncthreads();
  for (int off = 32; off > 0; off >>= 1) {
    if (t < off) red[t] = fmaxf(red[t], red[t + off]);
    __syncthreads();
  }
  float mx = red[0]; __syncthreads();
  float e = expf(v - mx);
  red[t] = e; __syncthreads();
  for (int off = 32; off > 0; off >>= 1) {
    if (t < off) red[t] += red[t + off];
    __syncthreads();
  }
  out[row * S_ + t] = e / red[0];
}

// semantic[b][r][d] = sum_s attn[b][r][s] * out[b][s][d]
__global__ void k_semantic(const float* __restrict__ attn,
                           const unsigned short* __restrict__ outbf,
                           unsigned short* __restrict__ sembf) {
  int idx = blockIdx.x * blockDim.x + threadIdx.x;
  if (idx >= B_ * R_ * 2 * H_) return;
  int d = idx % (2 * H_);
  int r = (idx / (2 * H_)) % R_;
  int b = idx / (2 * H_ * R_);
  const float* a = attn + ((long)b * R_ + r) * S_;
  const unsigned short* o = outbf + (long)b * S_ * (2 * H_) + d;
  float acc = 0.0f;
  for (int s = 0; s < S_; ++s) acc += a[s] * bf2f(o[(long)s * (2 * H_)]);
  sembf[idx] = f2bf(acc);
}

// Dynamic routing: one workgroup (256 thr = 8 waves) per batch element.
__global__ __launch_bounds__(256) void k_routing(
    const float* __restrict__ predOut, float* __restrict__ routesOut,
    float* __restrict__ clsOut)
{
  __shared__ float sLogits[R_ * C_];
  __shared__ float sRoutes[R_ * C_];
  __shared__ float sV[C_ * P_];
  const int b = blockIdx.x;
  const float* pg = predOut + (long)b * (R_ * C_ * P_);
  for (int i = threadIdx.x; i < R_ * C_; i += 256) sLogits[i] = 0.0f;
  __syncthreads();
  const int r = threadIdx.x >> 5, lane = threadIdx.x & 31;
  for (int it = 0; it < 3; ++it) {
    float lv[5]; float mx = -3.4e38f;
    for (int i = 0; i < 5; ++i) {
      int c = lane + 32 * i;
      lv[i] = (c < C_) ? sLogits[r * C_ + c] : -3.4e38f;
      mx = fmaxf(mx, lv[i]);
    }
    for (int off = 16; off > 0; off >>= 1) mx = fmaxf(mx, __shfl_xor(mx, off, 32));
    float sum = 0.0f;
    for (int i = 0; i < 5; ++i) {
      int c = lane + 32 * i;
      if (c < C_) { lv[i] = expf(lv[i] - mx); sum += lv[i]; }
    }
    for (int off = 16; off > 0; off >>= 1) sum += __shfl_xor(sum, off, 32);
    for (int i = 0; i < 5; ++i) {
      int c = lane + 32 * i;
      if (c < C_) sRoutes[r * C_ + c] = lv[i] / sum;
    }
    __syncthreads();
    if (threadIdx.x < C_) {
      int c = threadIdx.x;
      float vv[16]; float n2 = 0.0f;
      for (int p = 0; p < P_; ++p) {
        float s = 0.0f;
        for (int rr = 0; rr < R_; ++rr)
          s += sRoutes[rr * C_ + c] * pg[rr * (C_ * P_) + c * P_ + p];
        vv[p] = s; n2 += s * s;
      }
      float scale = (n2 / (1.0f + n2)) / sqrtf(n2 + 1e-9f);
      for (int p = 0; p < P_; ++p) sV[c * P_ + p] = scale * vv[p];
    }
    __syncthreads();
    for (int i = threadIdx.x; i < R_ * C_; i += 256) {
      int rr = i / C_, c = i % C_;
      float d = 0.0f;
      for (int p = 0; p < P_; ++p) d += pg[rr * (C_ * P_) + c * P_ + p] * sV[c * P_ + p];
      sLogits[i] += d;
    }
    __syncthreads();
  }
  for (int i = threadIdx.x; i < R_ * C_; i += 256)
    routesOut[(long)b * R_ * C_ + i] = sRoutes[i];
  if (threadIdx.x < C_) {
    int c = threadIdx.x; float n2 = 0.0f;
    for (int p = 0; p < P_; ++p) { float v = sV[c * P_ + p]; n2 += v * v; }
    clsOut[(long)b * C_ + c] = sqrtf(n2);
  }
}

// ---------------------------------------------------------------------------
extern "C" void kernel_launch(void* const* d_in, const int* in_sizes, int n_in,
                              void* d_out, int out_size, void* d_ws, size_t ws_size,
                              hipStream_t stream) {
  (void)in_sizes; (void)n_in; (void)out_size; (void)ws_size;
  const int*   input_x   = (const int*)d_in[0];
  const float* embedding = (const float*)d_in[2];
  const float* w_ih_f    = (const float*)d_in[3];
  const float* w_hh_f    = (const float*)d_in[4];
  const float* b_ih_f    = (const float*)d_in[5];
  const float* b_hh_f    = (const float*)d_in[6];
  const float* w_ih_b    = (const float*)d_in[7];
  const float* w_hh_b    = (const float*)d_in[8];
  const float* b_ih_b    = (const float*)d_in[9];
  const float* b_hh_b    = (const float*)d_in[10];
  const float* ws1       = (const float*)d_in[11];
  const float* ws2       = (const float*)d_in[12];
  const float* capsW     = (const float*)d_in[13];

  char* wp = (char*)d_ws;
  auto carve = [&](size_t bytes) -> char* {
    char* p = wp; wp += (bytes + 255) & ~(size_t)255; return p;
  };
  unsigned short* embbf = (unsigned short*)carve((size_t)S_ * B_ * EP_ * 2);       // [S][B][EP]
  unsigned short* wihbf = (unsigned short*)carve((size_t)2 * 4 * H_ * EP_ * 2);    // [dir][4H][EP]
  unsigned short* whhbf = (unsigned short*)carve((size_t)2 * 4 * H_ * H_ * 2);     // [dir][4H][H]
  unsigned short* ws1bf = (unsigned short*)carve((size_t)DAP_ * 2 * H_ * 2);       // [DAP][2H]
  unsigned short* capsT = (unsigned short*)carve((size_t)R_ * CPP_ * (2 * H_) * 2);// [r][CPP][2H]
  float*          zbuf  = (float*)carve((size_t)2 * B_ * 4 * H_ * 4);              // [dir][B][4H]
  float*          cbuf  = (float*)carve((size_t)2 * B_ * H_ * 4);
  unsigned short* hbf   = (unsigned short*)carve((size_t)2 * B_ * H_ * 2);
  unsigned short* outbf = (unsigned short*)carve((size_t)B_ * S_ * 2 * H_ * 2);    // [B][S][2H]
  unsigned short* prebf = (unsigned short*)carve((size_t)B_ * S_ * DA_ * 2);       // [B*S][DA]
  float*          attbuf = (float*)carve((size_t)B_ * R_ * S_ * 4);
  unsigned short* sembf = (unsigned short*)carve((size_t)B_ * R_ * 2 * H_ * 2);

  float* attnOut   = (float*)d_out;
  float* clsOut    = attnOut + (size_t)B_ * R_ * S_;
  float* predOut   = clsOut  + (size_t)B_ * C_;
  float* routesOut = predOut + (size_t)B_ * R_ * C_ * P_;

  hipMemsetAsync(cbuf, 0, (size_t)2 * B_ * H_ * 4, stream);
  hipMemsetAsync(hbf,  0, (size_t)2 * B_ * H_ * 2, stream);

  auto cvt2d = [&](const float* s, unsigned short* d, long rows, int K, int Kp) {
    long n = rows * Kp;
    k_cvt2d<<<dim3((unsigned)((n + 255) / 256)), 256, 0, stream>>>(s, d, rows, K, Kp);
  };
  cvt2d(w_ih_f, wihbf,                        4 * H_, E_, EP_);
  cvt2d(w_ih_b, wihbf + (long)4 * H_ * EP_,   4 * H_, E_, EP_);
  cvt2d(w_hh_f, whhbf,                        4 * H_, H_, H_);
  cvt2d(w_hh_b, whhbf + (long)4 * H_ * H_,    4 * H_, H_, H_);
  cvt2d(ws1,    ws1bf,                        DA_, 2 * H_, 2 * H_);
  k_cvtT<<<dim3((R_ * (2 * H_) * CP_ + 255) / 256), 256, 0, stream>>>(capsW, capsT);

  k_embed<<<dim3((S_ * B_ * EP_ + 255) / 256), 256, 0, stream>>>(input_x, embedding, embbf);

  // --- bidirectional LSTM: fused z = x@Wih^T + h@Whh^T per (step, dir) ---
  for (int t = 0; t < S_; ++t) {
    for (int dir = 0; dir < 2; ++dir) {
      int s = dir ? (S_ - 1 - t) : t;
      k_gemm<<<dim3(4 * H_ / 64, B_ / 64, 1), 128, 0, stream>>>(
          embbf + (long)s * B_ * EP_, EP_, 0,
          wihbf + (long)dir * 4 * H_ * EP_, EP_, 0,
          hbf + (long)dir * B_ * H_, H_,
          whhbf + (long)dir * 4 * H_ * H_, H_, H_,
          zbuf + (long)dir * B_ * 4 * H_, 4 * H_, 0,
          B_, 4 * H_, EP_, /*act=*/0);
    }
    k_gates<<<dim3((B_ * H_ + 255) / 256, 1, 2), 256, 0, stream>>>(
        zbuf, cbuf, hbf, outbf, b_ih_f, b_hh_f, b_ih_b, b_hh_b, t);
  }

  // --- pre = tanh(output @ ws1^T)  -> bf16 [B*S][DA] ---
  k_gemm<<<dim3(DAP_ / 64, (B_ * S_) / 64, 1), 128, 0, stream>>>(
      outbf, 2 * H_, 0,
      ws1bf, 2 * H_, 0,
      nullptr, 0, nullptr, 0, 0,
      (void*)prebf, DA_, 0,
      B_ * S_, DA_, 2 * H_, /*act=tanh->bf16*/ 1);

  k_att<<<dim3((B_ * R_ * S_ + 255) / 256), 256, 0, stream>>>(prebf, ws2, attbuf);
  k_softmax64<<<dim3(B_ * R_), 64, 0, stream>>>(attbuf, attnOut);
  k_semantic<<<dim3((B_ * R_ * 2 * H_ + 255) / 256), 256, 0, stream>>>(attnOut, outbf, sembf);

  // --- pred[b][r][n] = sum_k semantic[b][r][k] * capsT[r][n][k]; z-batched over r ---
  k_gemm<<<dim3(CPP_ / 64, B_ / 64, R_), 128, 0, stream>>>(
      sembf, (long)R_ * 2 * H_, (long)2 * H_,
      capsT, (long)2 * H_, (long)CPP_ * (2 * H_),
      nullptr, 0, nullptr, 0, 0,
      (void*)predOut, (long)R_ * C_ * P_, (long)C_ * P_,
      B_, C_ * P_, 2 * H_, /*act=*/0);

  k_routing<<<dim3(B_), 256, 0, stream>>>(predOut, routesOut, clsOut);
}
